// ItttLinear_48919677501653
// MI455X (gfx1250) — matrix operand order repaired and checked by reading, not compiled, not run on hardware
//
#include <hip/hip_runtime.h>
#include <math.h>

// Problem constants (match reference)
#define B_    4
#define S_    2048
#define IN_   2048
#define OUT_  2048
#define RANK_ 256
#define BASE_LR  0.01f
#define SCALER   45.25483399593904f   // sqrt(2048)

// GEMM tiling: 256 threads = 8 waves (2 M x 4 N), wave tile 64x64,
// block tile 128(M) x 256(N), K-step 32, double-buffered LDS staging.
#define BM 128
#define BN 256
#define BK 32
#define PA 40                 // padded LDS row stride (f16) = 32 + 8
#define PB 40
#define A_F16 (BM * PA)       // 5120 f16 per A buffer
#define BF16S (BN * PB)       // 10240 f16 per B buffer
#define SM_TOT (2 * A_F16 + 2 * BF16S)   // 30720 f16 = 61440 B LDS

typedef __attribute__((ext_vector_type(16))) _Float16 v16h;
typedef __attribute__((ext_vector_type(8)))  _Float16 v8h;
typedef __attribute__((ext_vector_type(4)))  _Float16 v4h;
typedef __attribute__((ext_vector_type(8)))  float    v8f;

// ---------------------------------------------------------------------------
// Elementwise prep kernels
// ---------------------------------------------------------------------------
__global__ __launch_bounds__(256) void cvt_f32_f16(const float* __restrict__ src,
                                                   _Float16* __restrict__ dst, int n) {
    int i = (blockIdx.x * blockDim.x + threadIdx.x) * 4;
    if (i < n) {
        float4 v = *(const float4*)(src + i);
        v4h h;
        h[0] = (_Float16)v.x; h[1] = (_Float16)v.y;
        h[2] = (_Float16)v.z; h[3] = (_Float16)v.w;
        *(v4h*)(dst + i) = h;
    }
}

__global__ __launch_bounds__(256) void make_s(const float* __restrict__ log_lr,
                                              const float* __restrict__ state,
                                              _Float16* __restrict__ s16, int n) {
    int i = (blockIdx.x * blockDim.x + threadIdx.x) * 4;
    if (i < n) {
        int ri = i % (RANK_ * IN_);
        float4 ll = *(const float4*)(log_lr + ri);
        float4 st = *(const float4*)(state + i);
        v4h h;
        h[0] = (_Float16)(BASE_LR * __expf(ll.x * SCALER) * st.x);
        h[1] = (_Float16)(BASE_LR * __expf(ll.y * SCALER) * st.y);
        h[2] = (_Float16)(BASE_LR * __expf(ll.z * SCALER) * st.z);
        h[3] = (_Float16)(BASE_LR * __expf(ll.w * SCALER) * st.w);
        *(v4h*)(s16 + i) = h;
    }
}

// ---------------------------------------------------------------------------
// Fragment helpers (CDNA5 v_wmma_f32_16x16x32_f16 layouts, ISA 7.12.2)
// ---------------------------------------------------------------------------
__device__ __forceinline__ v16h concat8(v8h lo, v8h hv) {
    return __builtin_shufflevector(lo, hv, 0,1,2,3,4,5,6,7,8,9,10,11,12,13,14,15);
}

// Direct-from-global A fragment (row = lane&15; lanes 0-15: K{0..7,16..23})
__device__ __forceinline__ v16h gload_frag_a(const _Float16* __restrict__ A, int lda,
                                             int row, int k0, int hi) {
    const _Float16* p = A + (size_t)row * lda + k0 + hi * 8;
    return concat8(*(const v8h*)p, *(const v8h*)(p + 16));
}

// Direct-from-global B fragment (col = lane&15 row of (out,in) matrix; lanes 0-15: K{0..7,16..23})
__device__ __forceinline__ v16h gload_frag_b(const _Float16* __restrict__ W, int ldb,
                                             int col, int k0, int hi) {
    const _Float16* p = W + (size_t)col * ldb + k0 + hi * 8;
    return concat8(*(const v8h*)p, *(const v8h*)(p + 16));
}

__device__ __forceinline__ v8f wmma_f16(v16h a, v16h b, v8f c) {
    return __builtin_amdgcn_wmma_f32_16x16x32_f16(false, a, false, b, (short)0, c,
                                                  false, false);
}

// ---------------------------------------------------------------------------
// Async staging: fill one K-step tile (A: BMx32, B: BNx32) into LDS using
// CDNA5 global_load_async_to_lds_b128 (ASYNCcnt-tracked, GVS addressing).
// 16 bytes (8 f16) per issue per lane; 2 rounds for A, 4 rounds for B.
// ---------------------------------------------------------------------------
__device__ __forceinline__ void async_stage(unsigned long long baseA,
                                            unsigned long long baseB,
                                            unsigned ldA, unsigned ldB, unsigned kt,
                                            unsigned ldsA, unsigned ldsB) {
    const unsigned t = threadIdx.x;
#pragma unroll
    for (unsigned rnd = 0; rnd < (BM * BK) / (8u * 256u); ++rnd) {   // 2 rounds
        unsigned c = t + rnd * 256u;
        unsigned r = c >> 2, kc = (c & 3u) * 8u;
        unsigned lds  = ldsA + (r * PA + kc) * 2u;
        unsigned voff = (r * ldA + kt + kc) * 2u;
        asm volatile("global_load_async_to_lds_b128 %0, %1, %2"
                     :: "v"(lds), "v"(voff), "s"(baseA) : "memory");
    }
#pragma unroll
    for (unsigned rnd = 0; rnd < (BN * BK) / (8u * 256u); ++rnd) {   // 4 rounds
        unsigned c = t + rnd * 256u;
        unsigned r = c >> 2, kc = (c & 3u) * 8u;
        unsigned lds  = ldsB + (r * PB + kc) * 2u;
        unsigned voff = (r * ldB + kt + kc) * 2u;
        asm volatile("global_load_async_to_lds_b128 %0, %1, %2"
                     :: "v"(lds), "v"(voff), "s"(baseB) : "memory");
    }
}

__device__ __forceinline__ void wait_async0() {
    asm volatile("s_wait_asynccnt 0x0" ::: "memory");
}

// ---------------------------------------------------------------------------
// z16[b] = x16[b] (SxIN) @ s16[b]^T (RANKxIN) -> SxRANK (f16)
// ---------------------------------------------------------------------------
__global__ __launch_bounds__(256) void kernel_z(const _Float16* __restrict__ x16,
                                                const _Float16* __restrict__ s16,
                                                _Float16* __restrict__ z16) {
    __shared__ _Float16 smem[SM_TOT];
    const int lane = threadIdx.x & 31;
    const int wave = threadIdx.x >> 5;
    const int wm = wave >> 2, wn = wave & 3;
    const int b  = blockIdx.z;
    const int r16 = lane & 15, hi = lane >> 4;

    const _Float16* gA = x16 + (size_t)b * S_ * IN_ + (size_t)blockIdx.x * BM * IN_;
    const _Float16* gB = s16 + (size_t)b * RANK_ * IN_;    // BN == RANK_

    const unsigned sm0 = (unsigned)(uintptr_t)(&smem[0]);
    const unsigned oA0 = sm0, oA1 = sm0 + A_F16 * 2u;
    const unsigned oB0 = sm0 + 2u * A_F16 * 2u, oB1 = oB0 + BF16S * 2u;

    v8f acc[4][4];
#pragma unroll
    for (int t = 0; t < 4; ++t)
#pragma unroll
        for (int j = 0; j < 4; ++j) acc[t][j] = (v8f)(0.0f);

    async_stage((unsigned long long)gA, (unsigned long long)gB, IN_, IN_, 0, oA0, oB0);

    for (int it = 0; it < IN_ / BK; ++it) {
        const int cur = it & 1;
        wait_async0();
        __syncthreads();
        if (it + 1 < IN_ / BK)
            async_stage((unsigned long long)gA, (unsigned long long)gB, IN_, IN_,
                        (unsigned)((it + 1) * BK),
                        cur ? oA0 : oA1, cur ? oB0 : oB1);

        const _Float16* sA = &smem[cur ? A_F16 : 0];
        const _Float16* sB = &smem[2 * A_F16 + (cur ? BF16S : 0)];

        v16h af[4];
#pragma unroll
        for (int t = 0; t < 4; ++t) {
            const _Float16* p = sA + (wm * 64 + t * 16 + r16) * PA + hi * 8;
            af[t] = concat8(*(const v8h*)p, *(const v8h*)(p + 16));
        }
#pragma unroll
        for (int j = 0; j < 4; ++j) {
            const _Float16* p = sB + (wn * 64 + j * 16 + r16) * PB + hi * 8;
            v16h bf = concat8(*(const v8h*)p, *(const v8h*)(p + 16));
#pragma unroll
            for (int t = 0; t < 4; ++t) acc[t][j] = wmma_f16(af[t], bf, acc[t][j]);
        }
    }

    _Float16* Z = z16 + (size_t)b * S_ * RANK_;
    const int mBase = blockIdx.x * BM + wm * 64;
#pragma unroll
    for (int t = 0; t < 4; ++t)
#pragma unroll
        for (int j = 0; j < 4; ++j)
#pragma unroll
            for (int v = 0; v < 8; ++v) {
                int r = mBase + t * 16 + hi * 8 + v;
                int c = wn * 64 + j * 16 + r16;
                Z[(size_t)r * RANK_ + c] = (_Float16)acc[t][j][v];
            }
}

// ---------------------------------------------------------------------------
// y[b] = x16[b] @ w16^T (K=IN, LDS-staged) + z16[b] @ p16^T (K=RANK) + bias
// ---------------------------------------------------------------------------
__global__ __launch_bounds__(256) void kernel_y(const _Float16* __restrict__ x16,
                                                const _Float16* __restrict__ w16,
                                                const _Float16* __restrict__ z16,
                                                const _Float16* __restrict__ p16,
                                                const float* __restrict__ bias,
                                                float* __restrict__ y) {
    __shared__ _Float16 smem[SM_TOT];
    const int lane = threadIdx.x & 31;
    const int wave = threadIdx.x >> 5;
    const int wm = wave >> 2, wn = wave & 3;
    const int b  = blockIdx.z;
    const int r16 = lane & 15, hi = lane >> 4;

    const _Float16* gA = x16 + (size_t)b * S_ * IN_ + (size_t)blockIdx.x * BM * IN_;
    const _Float16* gB = w16 + (size_t)blockIdx.y * BN * IN_;

    const unsigned sm0 = (unsigned)(uintptr_t)(&smem[0]);
    const unsigned oA0 = sm0, oA1 = sm0 + A_F16 * 2u;
    const unsigned oB0 = sm0 + 2u * A_F16 * 2u, oB1 = oB0 + BF16S * 2u;

    v8f acc[4][4];
#pragma unroll
    for (int t = 0; t < 4; ++t)
#pragma unroll
        for (int j = 0; j < 4; ++j) acc[t][j] = (v8f)(0.0f);

    async_stage((unsigned long long)gA, (unsigned long long)gB, IN_, IN_, 0, oA0, oB0);

    for (int it = 0; it < IN_ / BK; ++it) {
        const int cur = it & 1;
        wait_async0();
        __syncthreads();
        if (it + 1 < IN_ / BK)
            async_stage((unsigned long long)gA, (unsigned long long)gB, IN_, IN_,
                        (unsigned)((it + 1) * BK),
                        cur ? oA0 : oA1, cur ? oB0 : oB1);

        const _Float16* sA = &smem[cur ? A_F16 : 0];
        const _Float16* sB = &smem[2 * A_F16 + (cur ? BF16S : 0)];

        v16h af[4];
#pragma unroll
        for (int t = 0; t < 4; ++t) {
            const _Float16* p = sA + (wm * 64 + t * 16 + r16) * PA + hi * 8;
            af[t] = concat8(*(const v8h*)p, *(const v8h*)(p + 16));
        }
#pragma unroll
        for (int j = 0; j < 4; ++j) {
            const _Float16* p = sB + (wn * 64 + j * 16 + r16) * PB + hi * 8;
            v16h bf = concat8(*(const v8h*)p, *(const v8h*)(p + 16));
#pragma unroll
            for (int t = 0; t < 4; ++t) acc[t][j] = wmma_f16(af[t], bf, acc[t][j]);
        }
    }

    // LoRA pass: K = RANK_ = 256, direct global fragments (8 iterations)
    const _Float16* Zl = z16 + (size_t)b * S_ * RANK_;
    const int m0w = blockIdx.x * BM + wm * 64;
    const int n0w = blockIdx.y * BN + wn * 64;
    for (int k = 0; k < RANK_; k += BK) {
        v16h af[4];
#pragma unroll
        for (int t = 0; t < 4; ++t)
            af[t] = gload_frag_a(Zl, RANK_, m0w + t * 16 + r16, k, hi);
#pragma unroll
        for (int j = 0; j < 4; ++j) {
            v16h bf = gload_frag_b(p16, RANK_, n0w + j * 16 + r16, k, hi);
#pragma unroll
            for (int t = 0; t < 4; ++t) acc[t][j] = wmma_f16(af[t], bf, acc[t][j]);
        }
    }

    float* Y = y + (size_t)b * S_ * OUT_;
#pragma unroll
    for (int t = 0; t < 4; ++t)
#pragma unroll
        for (int j = 0; j < 4; ++j) {
            int c = n0w + j * 16 + r16;
            float bv = bias[c];
#pragma unroll
            for (int v = 0; v < 8; ++v) {
                int r = m0w + t * 16 + hi * 8 + v;
                Y[(size_t)r * OUT_ + c] = acc[t][j][v] + bv;
            }
        }
}

// ---------------------------------------------------------------------------
extern "C" void kernel_launch(void* const* d_in, const int* in_sizes, int n_in,
                              void* d_out, int out_size, void* d_ws, size_t ws_size,
                              hipStream_t stream) {
    const float* x        = (const float*)d_in[0];
    const float* weight   = (const float*)d_in[1];
    const float* bias     = (const float*)d_in[2];
    const float* log_lr   = (const float*)d_in[3];
    const float* out_proj = (const float*)d_in[4];
    const float* state    = (const float*)d_in[5];
    // d_in[6] = momentum, unused by forward

    char* ws = (char*)d_ws;
    _Float16* x16 = (_Float16*)ws;  ws += (size_t)B_ * S_ * IN_ * 2;
    _Float16* w16 = (_Float16*)ws;  ws += (size_t)OUT_ * IN_ * 2;
    _Float16* p16 = (_Float16*)ws;  ws += (size_t)OUT_ * RANK_ * 2;
    _Float16* s16 = (_Float16*)ws;  ws += (size_t)B_ * RANK_ * IN_ * 2;
    _Float16* z16 = (_Float16*)ws;  ws += (size_t)B_ * S_ * RANK_ * 2;

    const int nx = B_ * S_ * IN_;
    const int nw = OUT_ * IN_;
    const int np = OUT_ * RANK_;
    const int ns = B_ * RANK_ * IN_;

    cvt_f32_f16<<<nx / 1024, 256, 0, stream>>>(x, x16, nx);
    cvt_f32_f16<<<nw / 1024, 256, 0, stream>>>(weight, w16, nw);
    cvt_f32_f16<<<np / 1024, 256, 0, stream>>>(out_proj, p16, np);
    make_s    <<<ns / 1024, 256, 0, stream>>>(log_lr, state, s16, ns);

    kernel_z<<<dim3(S_ / BM, 1, B_), 256, 0, stream>>>(x16, s16, z16);
    kernel_y<<<dim3(S_ / BM, OUT_ / BN, B_), 256, 0, stream>>>(x16, w16, z16, p16,
                                                               bias, (float*)d_out);
}
